// EnhancedDDGAttention_33191507263871
// MI455X (gfx1250) — matrix operand order repaired and checked by reading, not compiled, not run on hardware
//
#include <hip/hip_runtime.h>
#include <hip/hip_bf16.h>

typedef __attribute__((ext_vector_type(16))) _Float16 v16h;
typedef __attribute__((ext_vector_type(8)))  _Float16 v8h;
typedef __attribute__((ext_vector_type(8)))  float    v8f;

#define N_B    8
#define L_S    512
#define D_IN   256
#define N_H    16
#define D_FEAT 640
#define D_HID  512
#define D_OUT  256
#define INF_V  100000.0f

__device__ __forceinline__ v8f vzero8() {
  v8f z = {0.f,0.f,0.f,0.f,0.f,0.f,0.f,0.f};
  return z;
}

__device__ __forceinline__ v8f wmma16(v16h a, v16h b, v8f c) {
  return __builtin_amdgcn_wmma_f32_16x16x32_f16(false, a, false, b, (short)0, c,
                                                false, false);
}

// Async 16B global -> LDS copy (CDNA5 GLOBAL_LOAD_ASYNC_TO_LDS_B128, ASYNCcnt).
__device__ __forceinline__ void async_b128(void* lds, const void* g) {
  unsigned l = (unsigned)(size_t)lds;
  unsigned long long ga = (unsigned long long)(size_t)g;
  asm volatile("global_load_async_to_lds_b128 %0, %1, off"
               :: "v"(l), "v"(ga) : "memory");
}
__device__ __forceinline__ void wait_async() {
  asm volatile("s_wait_asynccnt 0x0" ::: "memory");
}

// A fragment (16x32 f16) from a row-major tile, leading dim `ld` (halves).
__device__ __forceinline__ v16h frag_a(const _Float16* base, int ld) {
  const int lane = threadIdx.x & 31;
  const _Float16* p = base + (lane & 15) * ld + ((lane >> 4) << 3);
  v8h lo = *(const v8h*)(p);
  v8h hi = *(const v8h*)(p + 16);
  v16h a;
#pragma unroll
  for (int i = 0; i < 8; ++i) { a[i] = lo[i]; a[i + 8] = hi[i]; }
  return a;
}

// B fragment (32x16 f16, KxN) from an N-major tile: row n holds contiguous K.
__device__ __forceinline__ v16h frag_b(const _Float16* base, int ld) {
  const int lane = threadIdx.x & 31;
  const _Float16* p = base + (lane & 15) * ld + ((lane >> 4) << 3);
  v8h lo = *(const v8h*)(p);
  v8h hi = *(const v8h*)(p + 16);
  v16h b;
#pragma unroll
  for (int i = 0; i < 8; ++i) { b[i] = lo[i]; b[i + 8] = hi[i]; }
  return b;
}

// ---------------------------------------------------------------------------
// Kernel 0: one-shot f32 -> f16 conversion; weights stored N-major (col,K)
// so every later tile stage is a contiguous 16B-aligned async copy.
// ---------------------------------------------------------------------------
__global__ __launch_bounds__(256) void convert_kernel(
    const float* __restrict__ x, const float* __restrict__ Wq,
    const float* __restrict__ Wk, const float* __restrict__ Wv,
    const float* __restrict__ W1, const float* __restrict__ W2,
    _Float16* __restrict__ xh, _Float16* __restrict__ WqT,
    _Float16* __restrict__ WkT, _Float16* __restrict__ WvT,
    _Float16* __restrict__ W1T, _Float16* __restrict__ W2T) {
  const int which = blockIdx.y;
  const int stride = gridDim.x * 256;
  const int i0 = blockIdx.x * 256 + threadIdx.x;
  if (which == 0) {
    for (int i = i0; i < 4096 * 256; i += stride) xh[i] = (_Float16)x[i];
  } else if (which <= 3) {
    const float* W = (which == 1) ? Wq : (which == 2) ? Wk : Wv;
    _Float16* T = (which == 1) ? WqT : (which == 2) ? WkT : WvT;
    for (int i = i0; i < 512 * 256; i += stride) {
      int col = i >> 8, k = i & 255;
      T[i] = (_Float16)W[k * 512 + col];
    }
  } else if (which == 4) {
    for (int i = i0; i < 512 * 640; i += stride) {
      int col = i / 640, k = i - col * 640;
      W1T[i] = (_Float16)W1[k * 512 + col];
    }
  } else {
    for (int i = i0; i < 256 * 512; i += stride) {
      int col = i >> 9, k = i & 511;
      W2T[i] = (_Float16)W2[k * 256 + col];
    }
  }
}

// ---------------------------------------------------------------------------
// Kernel 1: Q/K/V projections; tiles staged by async DMA, spat fused into Q.
// ---------------------------------------------------------------------------
__global__ __launch_bounds__(256) void proj_kernel(
    const _Float16* __restrict__ xh, const _Float16* __restrict__ WqT,
    const _Float16* __restrict__ WkT, const _Float16* __restrict__ WvT,
    const float* __restrict__ pCA, const float* __restrict__ seW1,
    const float* __restrict__ seb1, const float* __restrict__ seW2,
    const float* __restrict__ seb2, _Float16* __restrict__ Qh,
    _Float16* __restrict__ Kh, _Float16* __restrict__ VT) {
  const int which = blockIdx.z;
  const _Float16* WT = (which == 0) ? WqT : (which == 1) ? WkT : WvT;
  const int M0 = blockIdx.y * 64;
  const int N0 = blockIdx.x * 64;
  const int tid = threadIdx.x;
  const int wid = tid >> 5, lane = tid & 31;
  const int msub = wid & 1;
  const int ncol = wid >> 1;

  __shared__ __attribute__((aligned(32))) _Float16 As[64 * 32];
  __shared__ __attribute__((aligned(32))) _Float16 Bs[64 * 32];
  __shared__ float hidS[64 * 32];
  __shared__ float seW2S[32 * 64];

  if (which == 0) {
#pragma unroll
    for (int i = 0; i < 8; ++i) {
      int idx = i * 256 + tid;
      { int r = idx >> 5, j = idx & 31;
        const float* p = pCA + (size_t)(M0 + r) * 3;
        float t = p[0] * seW1[j] + p[1] * seW1[32 + j] + p[2] * seW1[64 + j] +
                  seb1[j];
        hidS[idx] = fmaxf(t, 0.f); }
      { int j = idx >> 6, c = idx & 63;
        seW2S[idx] = seW2[j * 512 + N0 + c]; }
    }
  }

  const int rA = tid >> 2, segA = tid & 3;
  v8f acc0 = vzero8(), acc1 = vzero8();
  for (int k0 = 0; k0 < D_IN; k0 += 32) {
    __syncthreads();
    async_b128(&As[rA * 32 + segA * 8],
               xh + (size_t)(M0 + rA) * 256 + k0 + segA * 8);
    async_b128(&Bs[rA * 32 + segA * 8],
               WT + (size_t)(N0 + rA) * 256 + k0 + segA * 8);
    wait_async();
    __syncthreads();
    v16h b  = frag_b(&Bs[(ncol * 16) * 32], 32);
    v16h a0 = frag_a(&As[(32 * msub) * 32], 32);
    v16h a1 = frag_a(&As[(32 * msub + 16) * 32], 32);
    acc0 = wmma16(a0, b, acc0);
    acc1 = wmma16(a1, b, acc1);
  }

#pragma unroll
  for (int half = 0; half < 2; ++half) {
    v8f acc = half ? acc1 : acc0;
    int rbase = 32 * msub + half * 16 + ((lane >> 4) << 3);
    int ncl = ncol * 16 + (lane & 15);
    int gcol = N0 + ncl;
#pragma unroll
    for (int v = 0; v < 8; ++v) {
      int r = rbase + v;
      int grow = M0 + r;
      float val = acc[v];
      if (which == 0) {
        float s = seb2[gcol];
        const float* hp = &hidS[r * 32];
#pragma unroll 8
        for (int j = 0; j < 32; ++j) s += hp[j] * seW2S[j * 64 + ncl];
        Qh[(size_t)grow * 512 + gcol] = (_Float16)(val + s);
      } else if (which == 1) {
        Kh[(size_t)grow * 512 + gcol] = (_Float16)val;
      } else {
        int n = grow >> 9, l = grow & 511;
        int hh = gcol >> 5, d = gcol & 31;
        VT[(size_t)(((n * N_H) + hh) * 32 + d) * 512 + l] = (_Float16)val;
      }
    }
  }
}

// ---------------------------------------------------------------------------
// Kernel 2: attention core. One block per (n, h, 16 queries).
// ---------------------------------------------------------------------------
__global__ __launch_bounds__(256) void attn_kernel(
    const _Float16* __restrict__ Qh, const _Float16* __restrict__ Kh,
    const _Float16* __restrict__ VT, const float* __restrict__ pCA,
    const float* __restrict__ pCB, const float* __restrict__ frame,
    const unsigned char* __restrict__ mask, _Float16* __restrict__ feat) {
  const int n = blockIdx.z, h = blockIdx.y, l0 = blockIdx.x * 16;
  const int tid = threadIdx.x, wid = tid >> 5, lane = tid & 31;

  __shared__ float logitsS[16 * 512];
  __shared__ __attribute__((aligned(32))) _Float16 alphaS[16 * 512];
  __shared__ float redS[16 * 16];
  __shared__ float apbP[16 * 8 * 4];
  __shared__ unsigned char mrowS[16];

  if (tid < 16) mrowS[tid] = mask[n * L_S + l0 + tid];

  const _Float16* Qbase = Qh + (size_t)(n * L_S + l0) * 512 + h * 32;
  v16h a = frag_a(Qbase, 512);
  const int kk0 = wid * 64;
  __syncthreads();
  if (wid == 6) {
    const char* vb = (const char*)(VT + (size_t)((n * N_H + h) * 32) * 512);
#pragma unroll
    for (int i = 0; i < 8; ++i)
      __builtin_prefetch(vb + (i * 32 + lane) * 128, 0, 0);
  } else if (wid == 7) {
    const char* cb = (const char*)(pCB + (size_t)n * L_S * 3);
#pragma unroll
    for (int i = 0; i < 2; ++i)
      __builtin_prefetch(cb + (i * 32 + lane) * 128, 0, 0);
  }
#pragma unroll
  for (int c = 0; c < 4; ++c) {
    const _Float16* Kbase =
        Kh + (size_t)(n * L_S + kk0 + c * 16) * 512 + h * 32;
    v16h b = frag_b(Kbase, 512);
    v8f acc = wmma16(a, b, vzero8());
    int key = kk0 + c * 16 + (lane & 15);
    unsigned char mk = mask[n * L_S + key];
#pragma unroll
    for (int v = 0; v < 8; ++v) {
      int r = ((lane >> 4) << 3) + v;
      float val = acc[v];
      if (!(mrowS[r] && mk)) val -= INF_V;
      logitsS[r * 512 + key] = val;
    }
  }
  __syncthreads();

  {
    int r = tid >> 4, sub = tid & 15;
    float* row = &logitsS[r * 512];
    float m = -3.0e38f;
    for (int i = 0; i < 32; ++i) m = fmaxf(m, row[sub * 32 + i]);
    redS[r * 16 + sub] = m;
    __syncthreads();
    float rm = redS[r * 16];
#pragma unroll
    for (int i = 1; i < 16; ++i) rm = fmaxf(rm, redS[r * 16 + i]);
    float tmp[32];
    float s = 0.f;
    for (int i = 0; i < 32; ++i) {
      float e = __expf(row[sub * 32 + i] - rm);
      tmp[i] = e; s += e;
    }
    __syncthreads();
    redS[r * 16 + sub] = s;
    __syncthreads();
    float rs = 0.f;
#pragma unroll
    for (int i = 0; i < 16; ++i) rs += redS[r * 16 + i];
    float inv = (mrowS[r] ? 1.f : 0.f) / rs;
    for (int i = 0; i < 32; ++i)
      alphaS[r * 512 + sub * 32 + i] = (_Float16)(tmp[i] * inv);
  }
  __syncthreads();

  if (wid < 2) {
    const int d0 = wid * 16;
    const _Float16* Vbase = VT + (size_t)(((n * N_H) + h) * 32 + d0) * 512;
    v8f acc = vzero8();
    for (int k0 = 0; k0 < 512; k0 += 32) {
      v16h aa = frag_a(&alphaS[k0], 512);
      v16h bb = frag_b(Vbase + k0, 512);
      acc = wmma16(aa, bb, acc);
    }
    int col = h * 32 + d0 + (lane & 15);
#pragma unroll
    for (int v = 0; v < 8; ++v) {
      int r = ((lane >> 4) << 3) + v;
      feat[(size_t)(n * L_S + l0 + r) * D_FEAT + col] = (_Float16)acc[v];
    }
  } else if (wid < 6) {
    int t = tid - 64;
    int r = t >> 3, part = t & 7;
    const float* cb = pCB + (size_t)n * L_S * 3;
    float s0 = 0.f, s1 = 0.f, s2 = 0.f, sa = 0.f;
    for (int k = part * 64; k < part * 64 + 64; ++k) {
      float al = (float)alphaS[r * 512 + k];
      sa += al;
      s0 += al * cb[k * 3 + 0];
      s1 += al * cb[k * 3 + 1];
      s2 += al * cb[k * 3 + 2];
    }
    float* p = &apbP[(r * 8 + part) * 4];
    p[0] = s0; p[1] = s1; p[2] = s2; p[3] = sa;
  }
  __syncthreads();

  if (tid < 16) {
    int r = tid;
    float s0 = 0.f, s1 = 0.f, s2 = 0.f, sa = 0.f;
#pragma unroll
    for (int part = 0; part < 8; ++part) {
      const float* p = &apbP[(r * 8 + part) * 4];
      s0 += p[0]; s1 += p[1]; s2 += p[2]; sa += p[3];
    }
    int gl = n * L_S + l0 + r;
    float a0 = s0 - pCA[gl * 3 + 0] * sa;
    float a1 = s1 - pCA[gl * 3 + 1] * sa;
    float a2 = s2 - pCA[gl * 3 + 2] * sa;
    float dist = sqrtf(a0 * a0 + a1 * a1 + a2 * a2);
    const float* F = frame + (size_t)gl * 9;
    float p0 = F[0] * a0 + F[1] * a1 + F[2] * a2;
    float p1 = F[3] * a0 + F[4] * a1 + F[5] * a2;
    float p2 = F[6] * a0 + F[7] * a1 + F[8] * a2;
    float nrm = sqrtf(p0 * p0 + p1 * p1 + p2 * p2) + 1e-10f;
    float ang = atan2f(p1, p0);
    _Float16* fr = feat + (size_t)gl * D_FEAT + 512;
    fr[h * 3 + 0] = (_Float16)p0;
    fr[h * 3 + 1] = (_Float16)p1;
    fr[h * 3 + 2] = (_Float16)p2;
    fr[48 + h] = (_Float16)dist;
    fr[64 + h * 3 + 0] = (_Float16)(p0 / nrm);
    fr[64 + h * 3 + 1] = (_Float16)(p1 / nrm);
    fr[64 + h * 3 + 2] = (_Float16)(p2 / nrm);
    fr[112 + h] = (_Float16)ang;
  }
}

// ---------------------------------------------------------------------------
// Kernel 3: output MLP + masked residual + layernorm; async-staged tiles.
// ---------------------------------------------------------------------------
__global__ __launch_bounds__(256) void mlp_ln_kernel(
    const _Float16* __restrict__ feat, const float* __restrict__ x,
    const _Float16* __restrict__ W1T, const float* __restrict__ b1,
    const _Float16* __restrict__ W2T, const float* __restrict__ b2,
    const float* __restrict__ ln_g, const float* __restrict__ ln_b,
    const unsigned char* __restrict__ mask, float* __restrict__ out) {
  const int M0 = blockIdx.x * 16;
  const int tid = threadIdx.x, wid = tid >> 5, lane = tid & 31;

  __shared__ __attribute__((aligned(32))) _Float16 aS[16 * 32];
  __shared__ __attribute__((aligned(32))) _Float16 h1S[16 * 512];
  __shared__ __attribute__((aligned(32))) float uS[8192];
  __shared__ float redS[16 * 16];
  _Float16* bS = (_Float16*)uS;
  float* outS = uS;

  v8f acc[4];
#pragma unroll
  for (int c = 0; c < 4; ++c) acc[c] = vzero8();
  for (int k0 = 0; k0 < D_FEAT; k0 += 32) {
    __syncthreads();
    if (tid < 64) {
      int r = tid >> 2, seg = tid & 3;
      async_b128(&aS[r * 32 + seg * 8],
                 feat + (size_t)(M0 + r) * D_FEAT + k0 + seg * 8);
    }
#pragma unroll
    for (int i = 0; i < 8; ++i) {
      int idx = i * 256 + tid;
      int col = idx >> 2, seg = idx & 3;
      async_b128(&bS[col * 32 + seg * 8],
                 W1T + (size_t)col * D_FEAT + k0 + seg * 8);
    }
    wait_async();
    __syncthreads();
    v16h aa = frag_a(aS, 32);
#pragma unroll
    for (int c = 0; c < 4; ++c) {
      v16h bb = frag_b(&bS[(wid * 64 + c * 16) * 32], 32);
      acc[c] = wmma16(aa, bb, acc[c]);
    }
  }
  __syncthreads();
#pragma unroll
  for (int c = 0; c < 4; ++c) {
    int col = wid * 64 + c * 16 + (lane & 15);
#pragma unroll
    for (int v = 0; v < 8; ++v) {
      int r = ((lane >> 4) << 3) + v;
      h1S[r * 512 + col] = (_Float16)fmaxf(acc[c][v] + b1[col], 0.f);
    }
  }

  v8f acc2[2];
#pragma unroll
  for (int c = 0; c < 2; ++c) acc2[c] = vzero8();
  for (int k0 = 0; k0 < D_HID; k0 += 32) {
    __syncthreads();
#pragma unroll
    for (int i = 0; i < 4; ++i) {
      int idx = i * 256 + tid;
      int col = idx >> 2, seg = idx & 3;
      async_b128(&bS[col * 32 + seg * 8],
                 W2T + (size_t)col * D_HID + k0 + seg * 8);
    }
    wait_async();
    __syncthreads();
    v16h aa = frag_a(&h1S[k0], 512);
#pragma unroll
    for (int c = 0; c < 2; ++c) {
      v16h bb = frag_b(&bS[(wid * 32 + c * 16) * 32], 32);
      acc2[c] = wmma16(aa, bb, acc2[c]);
    }
  }
  __syncthreads();
#pragma unroll
  for (int c = 0; c < 2; ++c) {
    int col = wid * 32 + c * 16 + (lane & 15);
#pragma unroll
    for (int v = 0; v < 8; ++v) {
      int r = ((lane >> 4) << 3) + v;
      outS[r * 256 + col] = acc2[c][v] + b2[col];
    }
  }
  __syncthreads();

  {
    int r = tid >> 4, sub = tid & 15;
    int grow = M0 + r;
    unsigned char mv = mask[grow];
    float yv[16];
    float s = 0.f, ss = 0.f;
#pragma unroll
    for (int i = 0; i < 16; ++i) {
      int col = sub * 16 + i;
      float o = mv ? outS[r * 256 + col] : 0.f;
      float y = x[(size_t)grow * 256 + col] + o;
      yv[i] = y; s += y; ss += y * y;
    }
    redS[r * 16 + sub] = s;
    __syncthreads();
    float mean = 0.f;
#pragma unroll
    for (int i = 0; i < 16; ++i) mean += redS[r * 16 + i];
    mean *= (1.f / 256.f);
    __syncthreads();
    redS[r * 16 + sub] = ss;
    __syncthreads();
    float m2 = 0.f;
#pragma unroll
    for (int i = 0; i < 16; ++i) m2 += redS[r * 16 + i];
    float var = m2 * (1.f / 256.f) - mean * mean;
    float rstd = rsqrtf(var + 1e-5f);
#pragma unroll
    for (int i = 0; i < 16; ++i) {
      int col = sub * 16 + i;
      out[(size_t)grow * 256 + col] =
          (yv[i] - mean) * rstd * ln_g[col] + ln_b[col];
    }
  }
}

extern "C" void kernel_launch(void* const* d_in, const int* in_sizes, int n_in,
                              void* d_out, int out_size, void* d_ws,
                              size_t ws_size, hipStream_t stream) {
  (void)in_sizes; (void)n_in; (void)out_size; (void)ws_size;
  const float* x     = (const float*)d_in[0];
  const float* pCA   = (const float*)d_in[1];
  const float* pCB   = (const float*)d_in[2];
  const float* frame = (const float*)d_in[3];
  const unsigned char* mask = (const unsigned char*)d_in[4];
  const float* Wq   = (const float*)d_in[5];
  const float* Wk   = (const float*)d_in[6];
  const float* Wv   = (const float*)d_in[7];
  const float* seW1 = (const float*)d_in[8];
  const float* seb1 = (const float*)d_in[9];
  const float* seW2 = (const float*)d_in[10];
  const float* seb2 = (const float*)d_in[11];
  const float* oW1  = (const float*)d_in[12];
  const float* ob1  = (const float*)d_in[13];
  const float* oW2  = (const float*)d_in[14];
  const float* ob2  = (const float*)d_in[15];
  const float* lng  = (const float*)d_in[16];
  const float* lnb  = (const float*)d_in[17];

  char* ws = (char*)d_ws;
  const size_t MB = 1024 * 1024;
  _Float16* Qh   = (_Float16*)(ws);
  _Float16* Kh   = (_Float16*)(ws + 4  * MB);
  _Float16* VT   = (_Float16*)(ws + 8  * MB);
  _Float16* feat = (_Float16*)(ws + 12 * MB);
  _Float16* xh   = (_Float16*)(ws + 18 * MB);
  _Float16* WqT  = (_Float16*)(ws + 20 * MB);
  _Float16* WkT  = (_Float16*)(ws + 20 * MB + 256 * 1024);
  _Float16* WvT  = (_Float16*)(ws + 20 * MB + 512 * 1024);
  _Float16* W1T  = (_Float16*)(ws + 21 * MB);
  _Float16* W2T  = (_Float16*)(ws + 22 * MB);

  convert_kernel<<<dim3(512, 6), 256, 0, stream>>>(
      x, Wq, Wk, Wv, oW1, oW2, xh, WqT, WkT, WvT, W1T, W2T);
  proj_kernel<<<dim3(8, 64, 3), 256, 0, stream>>>(
      xh, WqT, WkT, WvT, pCA, seW1, seb1, seW2, seb2, Qh, Kh, VT);
  attn_kernel<<<dim3(32, 16, 8), 256, 0, stream>>>(
      Qh, Kh, VT, pCA, pCB, frame, mask, feat);
  mlp_ln_kernel<<<dim3(256), 256, 0, stream>>>(
      feat, x, W1T, ob1, W2T, ob2, lng, lnb, mask, (float*)d_out);
}
